// MambaBlock_9912784519433
// MI455X (gfx1250) — compile-verified, matrix operand order checked by reading, not run on hardware
//
#include <hip/hip_runtime.h>
#include <hip/hip_bf16.h>
#include <stdint.h>

#define D_MODEL 1024
#define D_STATE 16
#define D_CONV  4
#define D_INNER 2048
#define DT_RANK 128
#define BATCH   2
#define SEQ     2048
#define ROWS    (BATCH * SEQ)            /* 4096 */
#define N_XBC   (DT_RANK + 2 * D_STATE)  /* 160  */

typedef __attribute__((ext_vector_type(16))) __bf16 v16bf;
typedef __attribute__((ext_vector_type(8)))  float  v8f;

union FragAB { v16bf v; uint4 q[2]; };

__device__ __forceinline__ unsigned short f2bf(float f) {
    unsigned int u = __float_as_uint(f);
    u += 0x7FFFu + ((u >> 16) & 1u);   // round-to-nearest-even
    return (unsigned short)(u >> 16);
}

// ---------------------------------------------------------------------------
// LayerNorm over D_MODEL=1024, one row per block, 256 threads x 4 elems.
// Emits bf16 normalized activations (A-operand of in_proj WMMA GEMM).
// ---------------------------------------------------------------------------
__global__ __launch_bounds__(256) void ln_bf16_kernel(
    const float* __restrict__ x, const float* __restrict__ w,
    const float* __restrict__ b, unsigned short* __restrict__ out)
{
    __shared__ float red[8];
    const int row = blockIdx.x;
    const int t   = threadIdx.x;
    const float* xr = x + (size_t)row * D_MODEL;
    float4 v = ((const float4*)xr)[t];

    float s = v.x + v.y + v.z + v.w;
    #pragma unroll
    for (int off = 16; off; off >>= 1) s += __shfl_xor(s, off, 32);
    if ((t & 31) == 0) red[t >> 5] = s;
    __syncthreads();
    float mu = (red[0] + red[1] + red[2] + red[3] +
                red[4] + red[5] + red[6] + red[7]) * (1.0f / D_MODEL);
    __syncthreads();

    float d0 = v.x - mu, d1 = v.y - mu, d2 = v.z - mu, d3 = v.w - mu;
    float s2 = d0 * d0 + d1 * d1 + d2 * d2 + d3 * d3;
    #pragma unroll
    for (int off = 16; off; off >>= 1) s2 += __shfl_xor(s2, off, 32);
    if ((t & 31) == 0) red[t >> 5] = s2;
    __syncthreads();
    float var = (red[0] + red[1] + red[2] + red[3] +
                 red[4] + red[5] + red[6] + red[7]) * (1.0f / D_MODEL);
    float rinv = rsqrtf(var + 1e-5f);

    float4 wv = ((const float4*)w)[t];
    float4 bv = ((const float4*)b)[t];
    ushort4 o;
    o.x = f2bf(d0 * rinv * wv.x + bv.x);
    o.y = f2bf(d1 * rinv * wv.y + bv.y);
    o.z = f2bf(d2 * rinv * wv.z + bv.z);
    o.w = f2bf(d3 * rinv * wv.w + bv.w);
    ((ushort4*)(out + (size_t)row * D_MODEL))[t] = o;
}

// ---------------------------------------------------------------------------
// f32 -> bf16 (optionally strided source): weights + the dt_raw slice of xBC.
// ---------------------------------------------------------------------------
__global__ __launch_bounds__(256) void f32_to_bf16_kernel(
    const float* __restrict__ src, unsigned short* __restrict__ dst,
    int rows, int cols, int ld)
{
    int idx = blockIdx.x * 256 + threadIdx.x;
    if (idx >= rows * cols) return;
    int r = idx / cols, c = idx - r * cols;
    dst[idx] = f2bf(src[(size_t)r * ld + c]);
}

// ---------------------------------------------------------------------------
// WMMA bf16 GEMM: out[M,N] = A[M,K] * W[N,K]^T (+bias)(+epilogue)
// Register-blocked: each wave32 computes MT x NT tiles of 16x16 (acc in f32).
// Per K-step: MT+NT fragment loads feed MT*NT v_wmma_f32_16x16x32_bf16 ops,
// so MT=NT=2 doubles FLOP/byte vs 1x1 and gives 4 independent WMMAs to hide
// XDL latency. Block = 8 waves (4 along M, 2 along N) -> (64*MT) x (32*NT).
// Fragment layouts follow CDNA5 ISA 7.12.2:
//   A: lane = 16*half + l16 holds row l16, K = k0+half*8+{0..7} and +16.
//   B: mirrored, columns from W rows (W is N x K row-major -> contiguous
//      16B dwordx4 loads per lane).
//   C/D: col = l16, rows = half*8 + v for VGPR v.
// mode: 0 = none, 1 = softplus (dt), 2 = add residual `res` (out_proj).
// ---------------------------------------------------------------------------
template<int MT, int NT>
__global__ __launch_bounds__(256) void gemm_bf16_wmma_kernel(
    const unsigned short* __restrict__ A,
    const unsigned short* __restrict__ W,
    const float* __restrict__ bias,
    const float* __restrict__ res,
    float* __restrict__ out,
    int M, int N, int K, int mode)
{
    const int lane = threadIdx.x & 31;
    const int wave = threadIdx.x >> 5;
    const int m0   = blockIdx.y * (64 * MT) + (wave & 3) * (16 * MT);
    const int n0   = blockIdx.x * (32 * NT) + (wave >> 2) * (16 * NT);
    const int half = lane >> 4;
    const int l16  = lane & 15;

    const unsigned short* arow[MT];
    const unsigned short* wrow[NT];
    #pragma unroll
    for (int mi = 0; mi < MT; ++mi)
        arow[mi] = A + (size_t)(m0 + mi * 16 + l16) * K + half * 8;
    #pragma unroll
    for (int ni = 0; ni < NT; ++ni)
        wrow[ni] = W + (size_t)(n0 + ni * 16 + l16) * K + half * 8;

    v8f acc[MT][NT] = {};
    for (int k0 = 0; k0 < K; k0 += 32) {
        FragAB a[MT], bm[NT];
        #pragma unroll
        for (int mi = 0; mi < MT; ++mi) {
            a[mi].q[0] = *(const uint4*)(arow[mi] + k0);
            a[mi].q[1] = *(const uint4*)(arow[mi] + k0 + 16);
            __builtin_prefetch(arow[mi] + k0 + 32, 0, 1);  // global_prefetch_b8
        }
        #pragma unroll
        for (int ni = 0; ni < NT; ++ni) {
            bm[ni].q[0] = *(const uint4*)(wrow[ni] + k0);
            bm[ni].q[1] = *(const uint4*)(wrow[ni] + k0 + 16);
            __builtin_prefetch(wrow[ni] + k0 + 32, 0, 1);
        }
        #pragma unroll
        for (int mi = 0; mi < MT; ++mi)
            #pragma unroll
            for (int ni = 0; ni < NT; ++ni)
                acc[mi][ni] = __builtin_amdgcn_wmma_f32_16x16x32_bf16(
                    false, a[mi].v, false, bm[ni].v, (short)0, acc[mi][ni],
                    false, false);
    }

    #pragma unroll
    for (int ni = 0; ni < NT; ++ni) {
        const int n  = n0 + ni * 16 + l16;
        const float bv = bias ? bias[n] : 0.f;
        #pragma unroll
        for (int mi = 0; mi < MT; ++mi) {
            #pragma unroll
            for (int i = 0; i < 8; ++i) {
                const int m = m0 + mi * 16 + half * 8 + i;
                float val = acc[mi][ni][i] + bv;
                if (mode == 1)      val = (val > 20.f) ? val : log1pf(__expf(val));
                else if (mode == 2) val += res[(size_t)m * N + n];
                out[(size_t)m * N + n] = val;
            }
        }
    }
}

// ---------------------------------------------------------------------------
// Depthwise causal conv (D_CONV=4 along L) + SiLU. Reads the x_h half of xz.
// Writes f32 (u for the scan) and bf16 (A-operand of x_proj GEMM).
// ---------------------------------------------------------------------------
__global__ __launch_bounds__(256) void conv_silu_kernel(
    const float* __restrict__ xz, const float* __restrict__ cw,
    const float* __restrict__ cb, float* __restrict__ u,
    unsigned short* __restrict__ ubf)
{
    int idx = blockIdx.x * 256 + threadIdx.x;
    if (idx >= ROWS * D_INNER) return;
    const int d   = idx & (D_INNER - 1);
    const int row = idx >> 11;           // b*SEQ + l
    const int l   = row & (SEQ - 1);

    float acc = cb[d];
    #pragma unroll
    for (int j = 0; j < D_CONV; ++j) {
        int lp = l + j - (D_CONV - 1);
        if (lp >= 0)
            acc += xz[(size_t)(row + j - (D_CONV - 1)) * (2 * D_INNER) + d]
                   * cw[d * D_CONV + j];
    }
    float sv = acc / (1.f + __expf(-acc));   // SiLU
    u[idx]   = sv;
    ubf[idx] = f2bf(sv);
}

// ---------------------------------------------------------------------------
// Selective scan: sequential over L, parallel over (b,d,s).
// 16 contiguous lanes = 16 states of one channel; h stays in a register;
// y reduction over states via 4 shfl_xor steps within the 16-lane group.
// ---------------------------------------------------------------------------
__global__ __launch_bounds__(256) void scan_kernel(
    const float* __restrict__ dt,    // ROWS x D_INNER
    const float* __restrict__ xbc,   // ROWS x N_XBC
    const float* __restrict__ u,     // ROWS x D_INNER
    const float* __restrict__ A_log, // D_INNER x D_STATE
    float* __restrict__ y)           // ROWS x D_INNER
{
    const int s  = threadIdx.x & 15;
    const int gi = blockIdx.x * 16 + (threadIdx.x >> 4);  // b*D_INNER + d
    const int b  = gi >> 11;
    const int d  = gi & (D_INNER - 1);

    const float Av = -__expf(A_log[d * D_STATE + s]);
    float h = 0.f;
    const size_t rbase = (size_t)b * SEQ;
    for (int l = 0; l < SEQ; ++l) {
        const size_t row = rbase + l;
        const float dtv = dt[row * D_INNER + d];
        const float uv  = u[row * D_INNER + d];
        const float Bv  = xbc[row * N_XBC + DT_RANK + s];
        const float Cv  = xbc[row * N_XBC + DT_RANK + D_STATE + s];
        h = __expf(dtv * Av) * h + dtv * Bv * uv;
        float p = h * Cv;
        p += __shfl_xor(p, 8, 32);
        p += __shfl_xor(p, 4, 32);
        p += __shfl_xor(p, 2, 32);
        p += __shfl_xor(p, 1, 32);
        if (s == 0) y[row * D_INNER + d] = p;
    }
}

// ---------------------------------------------------------------------------
// y = (yscan + u*D) * silu(z)  -> bf16 (A-operand of out_proj GEMM)
// ---------------------------------------------------------------------------
__global__ __launch_bounds__(256) void combine_kernel(
    const float* __restrict__ yscan, const float* __restrict__ u,
    const float* __restrict__ xz, const float* __restrict__ Dp,
    unsigned short* __restrict__ ybf)
{
    int idx = blockIdx.x * 256 + threadIdx.x;
    if (idx >= ROWS * D_INNER) return;
    const int row = idx >> 11;
    const int d   = idx & (D_INNER - 1);
    float yv = yscan[idx] + u[idx] * Dp[d];
    float z  = xz[(size_t)row * (2 * D_INNER) + D_INNER + d];
    float o  = yv * (z / (1.f + __expf(-z)));
    ybf[idx] = f2bf(o);
}

// ---------------------------------------------------------------------------

extern "C" void kernel_launch(void* const* d_in, const int* in_sizes, int n_in,
                              void* d_out, int out_size, void* d_ws, size_t ws_size,
                              hipStream_t stream) {
    (void)in_sizes; (void)n_in; (void)out_size; (void)ws_size;

    const float* x          = (const float*)d_in[0];
    const float* norm_w     = (const float*)d_in[1];
    const float* norm_b     = (const float*)d_in[2];
    const float* in_proj_w  = (const float*)d_in[3];
    const float* conv_w     = (const float*)d_in[4];
    const float* conv_b     = (const float*)d_in[5];
    const float* x_proj_w   = (const float*)d_in[6];
    const float* dt_proj_w  = (const float*)d_in[7];
    const float* dt_proj_b  = (const float*)d_in[8];
    const float* A_log      = (const float*)d_in[9];
    const float* D_param    = (const float*)d_in[10];
    const float* out_proj_w = (const float*)d_in[11];
    float* out = (float*)d_out;

    char* ws = (char*)d_ws;
    size_t off = 0;
    auto alloc = [&](size_t bytes) {
        char* p = ws + off;
        off = (off + bytes + 255) & ~(size_t)255;
        return p;
    };
    unsigned short* xn_bf   = (unsigned short*)alloc((size_t)ROWS * D_MODEL * 2);
    unsigned short* w_in_bf = (unsigned short*)alloc((size_t)2 * D_INNER * D_MODEL * 2);
    unsigned short* w_xp_bf = (unsigned short*)alloc((size_t)N_XBC * D_INNER * 2);
    unsigned short* w_dt_bf = (unsigned short*)alloc((size_t)D_INNER * DT_RANK * 2);
    unsigned short* w_ou_bf = (unsigned short*)alloc((size_t)D_MODEL * D_INNER * 2);
    float*          xz      = (float*)alloc((size_t)ROWS * 2 * D_INNER * 4);
    float*          u_f32   = (float*)alloc((size_t)ROWS * D_INNER * 4);
    unsigned short* xh_bf   = (unsigned short*)alloc((size_t)ROWS * D_INNER * 2);
    float*          xbc     = (float*)alloc((size_t)ROWS * N_XBC * 4);
    unsigned short* dtr_bf  = (unsigned short*)alloc((size_t)ROWS * DT_RANK * 2);
    float*          dt_f32  = (float*)alloc((size_t)ROWS * D_INNER * 4);
    float*          yscan   = (float*)alloc((size_t)ROWS * D_INNER * 4);
    unsigned short* y_bf    = (unsigned short*)alloc((size_t)ROWS * D_INNER * 2);

    auto cvt = [&](const float* s, unsigned short* dptr, int r, int c, int ld) {
        int n = r * c;
        f32_to_bf16_kernel<<<(n + 255) / 256, 256, 0, stream>>>(s, dptr, r, c, ld);
    };

    // 1) LayerNorm -> xn (bf16)
    ln_bf16_kernel<<<ROWS, 256, 0, stream>>>(x, norm_w, norm_b, xn_bf);

    // 2) Weight precision conversions (f32 -> bf16)
    cvt(in_proj_w,  w_in_bf, 2 * D_INNER, D_MODEL, D_MODEL);
    cvt(x_proj_w,   w_xp_bf, N_XBC,       D_INNER, D_INNER);
    cvt(dt_proj_w,  w_dt_bf, D_INNER,     DT_RANK, DT_RANK);
    cvt(out_proj_w, w_ou_bf, D_MODEL,     D_INNER, D_INNER);

    // 3) in_proj GEMM: xz[4096,4096] = xn @ in_proj_w^T   (2x2 WMMA tiles)
    gemm_bf16_wmma_kernel<2, 2>
        <<<dim3((2 * D_INNER) / 64, ROWS / 128), 256, 0, stream>>>(
        xn_bf, w_in_bf, nullptr, nullptr, xz, ROWS, 2 * D_INNER, D_MODEL, 0);

    // 4) depthwise conv + SiLU -> u (f32) and xh (bf16)
    conv_silu_kernel<<<(ROWS * D_INNER + 255) / 256, 256, 0, stream>>>(
        xz, conv_w, conv_b, u_f32, xh_bf);

    // 5) x_proj GEMM: xBC[4096,160] = xh @ x_proj_w^T     (1x1, N=160)
    gemm_bf16_wmma_kernel<1, 1>
        <<<dim3(N_XBC / 32, ROWS / 64), 256, 0, stream>>>(
        xh_bf, w_xp_bf, nullptr, nullptr, xbc, ROWS, N_XBC, D_INNER, 0);

    // 6) dt_raw slice -> bf16
    cvt(xbc, dtr_bf, ROWS, DT_RANK, N_XBC);

    // 7) dt_proj GEMM + softplus: dt[4096,2048]           (2x2 WMMA tiles)
    gemm_bf16_wmma_kernel<2, 2>
        <<<dim3(D_INNER / 64, ROWS / 128), 256, 0, stream>>>(
        dtr_bf, w_dt_bf, dt_proj_b, nullptr, dt_f32, ROWS, D_INNER, DT_RANK, 1);

    // 8) selective scan (sequential in L, parallel over b*d*s)
    scan_kernel<<<(BATCH * D_INNER) / 16, 256, 0, stream>>>(
        dt_f32, xbc, u_f32, A_log, yscan);

    // 9) y = (yscan + u*D) * silu(z) -> bf16
    combine_kernel<<<(ROWS * D_INNER + 255) / 256, 256, 0, stream>>>(
        yscan, u_f32, xz, D_param, y_bf);

    // 10) out_proj GEMM + residual: out[4096,1024]        (2x2 WMMA tiles)
    gemm_bf16_wmma_kernel<2, 2>
        <<<dim3(D_MODEL / 64, ROWS / 128), 256, 0, stream>>>(
        y_bf, w_ou_bf, nullptr, x, out, ROWS, D_MODEL, D_INNER, 2);
}
